// Int4Linear_76862734729399
// MI455X (gfx1250) — matrix operand order repaired and checked by reading, not compiled, not run on hardware
//
#include <hip/hip_runtime.h>

typedef __attribute__((ext_vector_type(16))) __bf16 v16bf;
typedef __attribute__((ext_vector_type(8)))  __bf16 bf16x8;
typedef __attribute__((ext_vector_type(8)))  float  v8f;

constexpr int Mdim = 8192;
constexpr int Kdim = 4096;
constexpr int Ndim = 11008;
constexpr int Gsz  = 128;

constexpr int BM = 128;
constexpr int BN = 128;
constexpr int BK = 64;
constexpr int NKT = Kdim / BK;       // 64 K-steps
constexpr int LDS_STRIDE = BK + 8;   // 72 bf16 = 144B rows, keeps 16B alignment

struct XRegs { float4 v[8]; };               // 32 fp32 activations per thread
struct QRegs { int4 q[4]; float s, zs; };    // 16 packed ints (32 int4 weights)

// ---- global -> register staging (no waits consumed here) -----------------
__device__ __forceinline__ XRegs load_x_regs(const float* __restrict__ x,
                                             long gxrow, int k0, int shalf) {
    XRegs r;
    const float4* xg = (const float4*)(x + gxrow * (long)Kdim + k0 + shalf * 32);
#pragma unroll
    for (int j = 0; j < 8; ++j) r.v[j] = xg[j];
    return r;
}

__device__ __forceinline__ QRegs load_q_regs(const int* __restrict__ qw,
                                             const float* __restrict__ scales,
                                             const float* __restrict__ zeros,
                                             long gwrow, int k0, int shalf) {
    QRegs r;
    const int4* qg = (const int4*)(qw + gwrow * (long)(Kdim / 2) + (k0 >> 1) + shalf * 16);
#pragma unroll
    for (int j = 0; j < 4; ++j) r.q[j] = qg[j];
    const int g = k0 / Gsz;                  // constant per tile (64 | 128)
    r.s  = scales[gwrow * (Kdim / Gsz) + g];
    r.zs = zeros [gwrow * (Kdim / Gsz) + g] * r.s;
    return r;
}

// ---- register -> LDS staging (fp32->bf16 convert / int4 dequant) ---------
__device__ __forceinline__ void store_x_lds(__bf16 (*sAb)[LDS_STRIDE],
                                            const XRegs& r, int srow, int shalf) {
#pragma unroll
    for (int j = 0; j < 4; ++j) {
        float4 a = r.v[2 * j + 0];
        float4 b = r.v[2 * j + 1];
        bf16x8 t;
        t[0] = (__bf16)a.x; t[1] = (__bf16)a.y;
        t[2] = (__bf16)a.z; t[3] = (__bf16)a.w;
        t[4] = (__bf16)b.x; t[5] = (__bf16)b.y;
        t[6] = (__bf16)b.z; t[7] = (__bf16)b.w;
        *(bf16x8*)&sAb[srow][shalf * 32 + j * 8] = t;
    }
}

__device__ __forceinline__ void store_q_lds(__bf16 (*sBb)[LDS_STRIDE],
                                            const QRegs& r, int srow, int shalf) {
    const float s = r.s, zs = r.zs;
#pragma unroll
    for (int j = 0; j < 4; ++j) {
        int4 q = r.q[j];
        bf16x8 t;
        int v;
        v = q.x; t[0] = (__bf16)((float)(v & 15) * s - zs);
                 t[1] = (__bf16)((float)((v >> 4) & 15) * s - zs);
        v = q.y; t[2] = (__bf16)((float)(v & 15) * s - zs);
                 t[3] = (__bf16)((float)((v >> 4) & 15) * s - zs);
        v = q.z; t[4] = (__bf16)((float)(v & 15) * s - zs);
                 t[5] = (__bf16)((float)((v >> 4) & 15) * s - zs);
        v = q.w; t[6] = (__bf16)((float)(v & 15) * s - zs);
                 t[7] = (__bf16)((float)((v >> 4) & 15) * s - zs);
        *(bf16x8*)&sBb[srow][shalf * 32 + j * 8] = t;
    }
}

// ---- one BK=64 tile of WMMA compute (2 k-substeps, 2x4 tiles per wave) ---
__device__ __forceinline__ void compute_tile(const __bf16 (*sAb)[LDS_STRIDE],
                                             const __bf16 (*sBb)[LDS_STRIDE],
                                             v8f acc[2][4],
                                             int waveM, int waveN, int l16, int lhi) {
#pragma unroll
    for (int ks = 0; ks < 2; ++ks) {
        const int kk = ks * 32;

        // A fragments: lane<16 holds K {0..7,16..23}, lane>=16 holds {8..15,24..31}
        v16bf Af[2];
#pragma unroll
        for (int i = 0; i < 2; ++i) {
            const int row = waveM * 32 + i * 16 + l16;
            const int kb  = kk + lhi * 8;
            bf16x8 lo = *(const bf16x8*)&sAb[row][kb];
            bf16x8 hi = *(const bf16x8*)&sAb[row][kb + 16];
#pragma unroll
            for (int e = 0; e < 8; ++e) { Af[i][e] = lo[e]; Af[i][e + 8] = hi[e]; }
        }

        // B fragments: lane<16 holds K {0..7,16..23}, lane>=16 holds {8..15,24..31}
        v16bf Bf[4];
#pragma unroll
        for (int j = 0; j < 4; ++j) {
            const int col = waveN * 64 + j * 16 + l16;
            const int kb  = kk + lhi * 8;
            bf16x8 lo = *(const bf16x8*)&sBb[col][kb];
            bf16x8 hi = *(const bf16x8*)&sBb[col][kb + 16];
#pragma unroll
            for (int e = 0; e < 8; ++e) { Bf[j][e] = lo[e]; Bf[j][e + 8] = hi[e]; }
        }

#pragma unroll
        for (int i = 0; i < 2; ++i)
#pragma unroll
            for (int j = 0; j < 4; ++j)
                acc[i][j] = __builtin_amdgcn_wmma_f32_16x16x32_bf16(
                    false, Af[i], false, Bf[j], (short)0, acc[i][j], false, false);
    }
}

__global__ __launch_bounds__(256)
void int4_gemm_wmma_bf16(const float* __restrict__ x,
                         const int*   __restrict__ qw,
                         const float* __restrict__ scales,
                         const float* __restrict__ zeros,
                         const float* __restrict__ bias,
                         float*       __restrict__ out)
{
    // double-buffered LDS: stage tile t+1 while computing tile t
    __shared__ __align__(16) __bf16 sA[2][BM][LDS_STRIDE];
    __shared__ __align__(16) __bf16 sB[2][BN][LDS_STRIDE];

    const int tid   = threadIdx.x;
    const int lane  = tid & 31;
    const int wid   = tid >> 5;       // 0..7
    const int l16   = lane & 15;
    const int lhi   = lane >> 4;      // 0 or 1 (half-wave)
    const int waveM = wid & 3;        // 4 waves along M (32 rows each)
    const int waveN = wid >> 2;       // 2 waves along N (64 cols each)

    const long tileN = (long)blockIdx.x * BN;
    const long tileM = (long)blockIdx.y * BM;

    // staging: 2 threads per row, each covers 32 k-values
    const int  srow  = tid >> 1;      // 0..127
    const int  shalf = tid & 1;       // 0..1
    const long gxrow = tileM + srow;
    const long gwrow = tileN + srow;

    v8f acc[2][4];
#pragma unroll
    for (int i = 0; i < 2; ++i)
#pragma unroll
        for (int j = 0; j < 4; ++j)
            acc[i][j] = (v8f)0.0f;

    // ---- prologue: stage tile 0 into buffer 0 ----------------------------
    {
        XRegs xr = load_x_regs(x, gxrow, 0, shalf);
        QRegs qr = load_q_regs(qw, scales, zeros, gwrow, 0, shalf);
        store_x_lds(sA[0], xr, srow, shalf);
        store_q_lds(sB[0], qr, srow, shalf);
    }
    __syncthreads();

    int buf = 0;
    // ---- pipelined main loop: one barrier per K-step ---------------------
    for (int kt = 0; kt < NKT - 1; ++kt) {
        const int knext = (kt + 1) * BK;

        // 1) kick off global loads for tile t+1 (latency hidden by WMMAs)
        XRegs xr = load_x_regs(x, gxrow, knext, shalf);
        QRegs qr = load_q_regs(qw, scales, zeros, gwrow, knext, shalf);

        // keep L2 ahead by one more tile
        if (kt + 2 < NKT) {
            __builtin_prefetch(x  + gxrow * (long)Kdim       + (knext + BK)        + shalf * 32, 0, 0);
            __builtin_prefetch(qw + gwrow * (long)(Kdim / 2) + ((knext + BK) >> 1) + shalf * 16, 0, 0);
        }

        // 2) compute tile t from current buffer
        compute_tile(sA[buf], sB[buf], acc, waveM, waveN, l16, lhi);

        // 3) convert/dequant and stage tile t+1 into the other buffer
        store_x_lds(sA[buf ^ 1], xr, srow, shalf);
        store_q_lds(sB[buf ^ 1], qr, srow, shalf);

        __syncthreads();
        buf ^= 1;
    }

    // ---- epilogue compute of last tile -----------------------------------
    compute_tile(sA[buf], sB[buf], acc, waveM, waveN, l16, lhi);

    // ---- epilogue: C/D layout -> global, fused bias ----------------------
#pragma unroll
    for (int j = 0; j < 4; ++j) {
        const long ncol = tileN + waveN * 64 + j * 16 + l16;
        const float bv  = bias[ncol];
#pragma unroll
        for (int i = 0; i < 2; ++i) {
            const long mbase = tileM + waveM * 32 + i * 16 + lhi * 8;
#pragma unroll
            for (int r = 0; r < 8; ++r) {
                out[(mbase + r) * (long)Ndim + ncol] = acc[i][j][r] + bv;
            }
        }
    }
}

extern "C" void kernel_launch(void* const* d_in, const int* in_sizes, int n_in,
                              void* d_out, int out_size, void* d_ws, size_t ws_size,
                              hipStream_t stream) {
    (void)in_sizes; (void)n_in; (void)out_size; (void)d_ws; (void)ws_size;
    const float* x      = (const float*)d_in[0];
    const int*   qw     = (const int*)  d_in[1];
    const float* scales = (const float*)d_in[2];
    const float* zeros  = (const float*)d_in[3];
    const float* bias   = (const float*)d_in[4];
    float*       out    = (float*)d_out;

    dim3 grid(Ndim / BN, Mdim / BM);   // 86 x 64 blocks
    dim3 block(256);
    int4_gemm_wmma_bf16<<<grid, block, 0, stream>>>(x, qw, scales, zeros, bias, out);
}
